// MoPModule_36421322670643
// MI455X (gfx1250) — matrix operand order repaired and checked by reading, not compiled, not run on hardware
//
#include <hip/hip_runtime.h>
#include <hip/hip_bf16.h>

// Problem constants (match the reference)
#define B_SZ   8192
#define IN_SZ  512
#define H_SZ   2048
#define OUT_SZ 512
#define E_SZ   8
#define RH_SZ  256

// Tile sizes
#define BM 128
#define BN 128
#define BK 32
#define LDSA_STRIDE 40   // bf16 elements per LDS row (80 bytes, 16B aligned, padded vs 32)
#define LDSB_STRIDE 40

typedef __attribute__((ext_vector_type(16))) __bf16 bf16x16;
typedef __attribute__((ext_vector_type(8)))  float  f32x8;

union FragU { bf16x16 v; uint4 q[2]; };

template <bool Bv> struct BoolC { static constexpr bool value = Bv; };

__device__ __forceinline__ int imin(int a, int b) { return a < b ? a : b; }

// fp32 -> bf16 bits with round-to-nearest-even
__device__ __forceinline__ unsigned f2bfbits(float f) {
    unsigned u = __float_as_uint(f);
    return ((u + 0x7FFFu + ((u >> 16) & 1u)) >> 16) & 0xFFFFu;
}
__device__ __forceinline__ __bf16 to_bf16(float f) {
    unsigned short h = (unsigned short)f2bfbits(f);
    return __builtin_bit_cast(__bf16, h);
}
__device__ __forceinline__ float bf2f(__bf16 b) {
    unsigned short h = __builtin_bit_cast(unsigned short, b);
    return __uint_as_float(((unsigned)h) << 16);
}

// 64-bit global address -> async copy of 16B into LDS at 32-bit lds offset.
__device__ __forceinline__ void async_b128_to_lds(unsigned lds_off, const void* gptr) {
    asm volatile("global_load_async_to_lds_b128 %0, %1, off"
                 :: "v"(lds_off), "v"((unsigned long long)(uintptr_t)gptr)
                 : "memory");
}
__device__ __forceinline__ void wait_asynccnt0() {
    asm volatile("s_wait_asynccnt 0" ::: "memory");
}

// ---------------------------------------------------------------------------
// One-shot weight prep: W fp32 [K,N] row-major -> Wt bf16 [N,K] (N-major).
// blockIdx.z = expert (stride K*N in / N*K out). 64x64 tiles through LDS so
// both the global reads and the global writes are coalesced.
// ---------------------------------------------------------------------------
__global__ __launch_bounds__(256) void transpose_cvt_kernel(
    const float* __restrict__ W, __bf16* __restrict__ Wt, int K, int N)
{
    const int e  = blockIdx.z;
    const float* Wp  = W  + (size_t)e * (size_t)K * (size_t)N;
    __bf16*      Wtp = Wt + (size_t)e * (size_t)N * (size_t)K;
    const int k0 = blockIdx.x * 64;
    const int n0 = blockIdx.y * 64;

    __shared__ __bf16 s[64][72];   // 72 = 64 + pad (row stride 144B, 16B aligned)

    const int tid = threadIdx.x;
    // read phase: 4 threads per k-row, 16 fp32 each (coalesced)
    {
        const int kr = tid >> 2, seg = tid & 3;
        const float* src = Wp + (size_t)(k0 + kr) * (size_t)N + n0 + seg * 16;
        float4 f0 = *(const float4*)(src + 0);
        float4 f1 = *(const float4*)(src + 4);
        float4 f2 = *(const float4*)(src + 8);
        float4 f3 = *(const float4*)(src + 12);
        unsigned pk[8];
        pk[0] = f2bfbits(f0.x) | (f2bfbits(f0.y) << 16);
        pk[1] = f2bfbits(f0.z) | (f2bfbits(f0.w) << 16);
        pk[2] = f2bfbits(f1.x) | (f2bfbits(f1.y) << 16);
        pk[3] = f2bfbits(f1.z) | (f2bfbits(f1.w) << 16);
        pk[4] = f2bfbits(f2.x) | (f2bfbits(f2.y) << 16);
        pk[5] = f2bfbits(f2.z) | (f2bfbits(f2.w) << 16);
        pk[6] = f2bfbits(f3.x) | (f2bfbits(f3.y) << 16);
        pk[7] = f2bfbits(f3.z) | (f2bfbits(f3.w) << 16);
        *(uint4*)&s[kr][seg * 16 + 0] = *(uint4*)&pk[0];
        *(uint4*)&s[kr][seg * 16 + 8] = *(uint4*)&pk[4];
    }
    __syncthreads();
    // write phase: 4 threads per n-row, 16 bf16 each (coalesced 128B per 4 lanes)
    {
        const int n = tid >> 2, seg = tid & 3;
        const int ks = seg * 16;
        unsigned pk[8];
        #pragma unroll
        for (int i = 0; i < 8; ++i) {
            unsigned lo = (unsigned)__builtin_bit_cast(unsigned short, s[ks + 2 * i + 0][n]);
            unsigned hi = (unsigned)__builtin_bit_cast(unsigned short, s[ks + 2 * i + 1][n]);
            pk[i] = lo | (hi << 16);
        }
        __bf16* dst = Wtp + (size_t)(n0 + n) * (size_t)K + k0 + ks;
        *(uint4*)(dst + 0) = *(uint4*)&pk[0];
        *(uint4*)(dst + 8) = *(uint4*)&pk[4];
    }
}

// ---------------------------------------------------------------------------
// Tiled WMMA GEMM:  C[M,N] = act( A_bf16[M,K] x Wt_bf16[N,K]^T + bias[N] )
//   - Wt is pre-converted bf16, N-major -> B tiles are byte-copyable
//   - A and B tiles: GLOBAL_LOAD_ASYNC_TO_LDS_B128 (ASYNCcnt), double-buffered
//   - per-expert segmentation (seg[e]..seg[e+1]) via blockIdx.z (seg==null: all)
//   - optional gather of A rows through gmap; output bf16 or fp32 scattered
// ---------------------------------------------------------------------------
template <bool ELU, bool GATHER_A, bool SCATTER_OUT>
__global__ __launch_bounds__(256) void gemm_bf16_wmma(
    const __bf16* __restrict__ A, const __bf16* __restrict__ Wt,
    const float* __restrict__ bias, __bf16* __restrict__ Cbf,
    float* __restrict__ Cf32, const int* __restrict__ seg,
    const int* __restrict__ gmap, int Mfull, int N, int K)
{
    const int e   = blockIdx.z;
    int off = 0, cnt = Mfull;
    if (seg) { off = seg[e]; cnt = seg[e + 1] - off; }
    const int m0 = blockIdx.x * BM;
    if (m0 >= cnt) return;
    const int n0 = blockIdx.y * BN;

    const __bf16* Wtp = Wt + (size_t)e * (size_t)N * (size_t)K;
    const float*  bp  = bias + (size_t)e * (size_t)N;

    __shared__ __bf16 sA[2][BM * LDSA_STRIDE];
    __shared__ __bf16 sB[2][BN * LDSB_STRIDE];

    const int tid = threadIdx.x;

    // ---- tile loaders: 2 threads per row, 32B per thread (2 async b128 each) ----
    const int t_row  = tid >> 1;
    const int t_half = tid & 1;
    const int a_loc  = imin(m0 + t_row, cnt - 1);
    const int a_src  = GATHER_A ? gmap[off + a_loc] : (off + a_loc);
    const __bf16* Aptr = A   + (size_t)a_src * (size_t)K + t_half * 16;
    const __bf16* Bptr = Wtp + (size_t)(n0 + t_row) * (size_t)K + t_half * 16;

    // ---- wave tiling: 8 waves, each 64(M) x 32(N) = 4x2 WMMA tiles ----
    const int wid   = tid >> 5;
    const int lane  = tid & 31;
    const int wm    = (wid & 1) * 64;
    const int wn    = (wid >> 1) * 32;
    const int lhalf = lane >> 4;
    const int lmod  = lane & 15;

    f32x8 acc[4][2] = {};

    auto issueAB = [&](int k0, int buf) {
        unsigned la = (unsigned)(uintptr_t)&sA[buf][t_row * LDSA_STRIDE + t_half * 16];
        unsigned lb = (unsigned)(uintptr_t)&sB[buf][t_row * LDSB_STRIDE + t_half * 16];
        async_b128_to_lds(la,      Aptr + k0);
        async_b128_to_lds(la + 16, Aptr + k0 + 8);
        async_b128_to_lds(lb,      Bptr + k0);
        async_b128_to_lds(lb + 16, Bptr + k0 + 8);
    };
    auto compute = [&](int buf) {
        // fragment layouts per CDNA5 ISA 7.12.2 (wave32)
        bf16x16 afrag[4], bfrag[2];
        #pragma unroll
        for (int mt = 0; mt < 4; ++mt) {
            const int m = wm + mt * 16 + lmod;
            FragU f;
            f.q[0] = *(const uint4*)&sA[buf][m * LDSA_STRIDE + 8 * lhalf];       // K=8h+0..7
            f.q[1] = *(const uint4*)&sA[buf][m * LDSA_STRIDE + 16 + 8 * lhalf];  // K=16+8h+0..7
            afrag[mt] = f.v;
        }
        #pragma unroll
        for (int nt = 0; nt < 2; ++nt) {
            const int n = wn + nt * 16 + lmod;
            FragU f;
            f.q[0] = *(const uint4*)&sB[buf][n * LDSB_STRIDE + 8 * lhalf];       // K=8h+0..7
            f.q[1] = *(const uint4*)&sB[buf][n * LDSB_STRIDE + 16 + 8 * lhalf];  // K=16+8h+0..7
            bfrag[nt] = f.v;
        }
        #pragma unroll
        for (int mt = 0; mt < 4; ++mt)
            #pragma unroll
            for (int nt = 0; nt < 2; ++nt)
                acc[mt][nt] = __builtin_amdgcn_wmma_f32_16x16x32_bf16(
                    false, afrag[mt], false, bfrag[nt], (short)0, acc[mt][nt],
                    false, false);
    };

    // ---- software-pipelined, double-buffered main loop (1 barrier / k-step) ----
    const int steps = K / BK;
    issueAB(0, 0);
    wait_asynccnt0();
    __syncthreads();

    for (int s = 0; s < steps; ++s) {
        const int cur = s & 1;
        if (s + 1 < steps) issueAB((s + 1) * BK, cur ^ 1);
        compute(cur);
        wait_asynccnt0();
        __syncthreads();
    }

    // ---- epilogue: bias (+ELU); full-tile fast path avoids per-row guards ----
    auto epilogue = [&](auto guard_c) {
        constexpr bool GUARD = decltype(guard_c)::value;
        #pragma unroll
        for (int nt = 0; nt < 2; ++nt) {
            const int cn = n0 + wn + nt * 16 + lmod;
            const float bb = bp[cn];
            #pragma unroll
            for (int mt = 0; mt < 4; ++mt) {
                #pragma unroll
                for (int r = 0; r < 8; ++r) {
                    const int lm = wm + mt * 16 + r + 8 * lhalf;  // C row = r + 8*half
                    const int gm = m0 + lm;
                    if (!GUARD || gm < cnt) {
                        float v = acc[mt][nt][r] + bb;
                        if (ELU) v = v > 0.f ? v : (__expf(v) - 1.f);
                        if (SCATTER_OUT) {
                            const int orow = gmap[off + gm];
                            Cf32[(size_t)orow * (size_t)N + cn] = v;
                        } else {
                            Cbf[(size_t)(off + gm) * (size_t)N + cn] = to_bf16(v);
                        }
                    }
                }
            }
        }
    };
    if (m0 + BM <= cnt) epilogue(BoolC<false>{});
    else                epilogue(BoolC<true>{});
}

// ---------------------------------------------------------------------------
// Router tail: logits = h1 @ rw2 + rb2 (N=8), argmax -> choice. Wave per row.
// ---------------------------------------------------------------------------
__global__ __launch_bounds__(256) void router_argmax(
    const __bf16* __restrict__ h1, const float* __restrict__ rw2,
    const float* __restrict__ rb2, int* __restrict__ choice)
{
    const int wid = threadIdx.x >> 5, lane = threadIdx.x & 31;
    const int row = blockIdx.x * 8 + wid;
    if (row >= B_SZ) return;
    float s[E_SZ] = {};
    for (int k = lane; k < RH_SZ; k += 32) {
        const float hv = bf2f(h1[(size_t)row * RH_SZ + k]);
        #pragma unroll
        for (int e = 0; e < E_SZ; ++e) s[e] += hv * rw2[k * E_SZ + e];
    }
    #pragma unroll
    for (int e = 0; e < E_SZ; ++e)
        #pragma unroll
        for (int o = 16; o > 0; o >>= 1) s[e] += __shfl_xor(s[e], o, 32);
    if (lane == 0) {
        int best = 0; float bv = s[0] + rb2[0];
        #pragma unroll
        for (int e = 1; e < E_SZ; ++e) {
            const float v = s[e] + rb2[e];
            if (v > bv) { bv = v; best = e; }   // first-max wins, matches argmax
        }
        choice[row] = best;
    }
}

// ---------------------------------------------------------------------------
// Routing bookkeeping: meta[0..7]=counts, meta[8..16]=seg, meta[17..24]=cursors
// ---------------------------------------------------------------------------
__global__ void zero_meta(int* meta) { if (threadIdx.x < 32) meta[threadIdx.x] = 0; }

__global__ void count_kernel(const int* __restrict__ choice, int* meta) {
    const int i = blockIdx.x * 256 + threadIdx.x;
    if (i < B_SZ) atomicAdd(&meta[choice[i]], 1);
}

__global__ void scan_kernel(int* meta) {
    if (threadIdx.x == 0) {
        int acc = 0;
        for (int e = 0; e < E_SZ; ++e) { meta[8 + e] = acc; acc += meta[e]; }
        meta[16] = acc;
    }
}

__global__ void scatter_kernel(const int* __restrict__ choice, int* meta,
                               int* __restrict__ idx) {
    const int i = blockIdx.x * 256 + threadIdx.x;
    if (i < B_SZ) {
        const int e = choice[i];
        const int p = atomicAdd(&meta[17 + e], 1);
        idx[meta[8 + e] + p] = i;
    }
}

// fp32 -> bf16 bulk convert (x)
__global__ void cvt_kernel(const float* __restrict__ in, __bf16* __restrict__ out,
                           int n) {
    const int i = (blockIdx.x * 256 + threadIdx.x) * 4;
    if (i < n) {
        const float4 v = *(const float4*)(in + i);
        unsigned p0 = f2bfbits(v.x) | (f2bfbits(v.y) << 16);
        unsigned p1 = f2bfbits(v.z) | (f2bfbits(v.w) << 16);
        uint2 o; o.x = p0; o.y = p1;
        *(uint2*)(out + i) = o;
    }
}

// ---------------------------------------------------------------------------
extern "C" void kernel_launch(void* const* d_in, const int* in_sizes, int n_in,
                              void* d_out, int out_size, void* d_ws, size_t ws_size,
                              hipStream_t stream)
{
    const float* x   = (const float*)d_in[0];
    const float* rw0 = (const float*)d_in[1];
    const float* rb0 = (const float*)d_in[2];
    const float* rw1 = (const float*)d_in[3];
    const float* rb1 = (const float*)d_in[4];
    const float* rw2 = (const float*)d_in[5];
    const float* rb2 = (const float*)d_in[6];
    const float* w1  = (const float*)d_in[7];
    const float* b1  = (const float*)d_in[8];
    const float* w2  = (const float*)d_in[9];
    const float* b2  = (const float*)d_in[10];
    const float* w3  = (const float*)d_in[11];
    const float* b3  = (const float*)d_in[12];
    float* out = (float*)d_out;

    char* ws = (char*)d_ws;
    size_t o = 0;
    auto alloc = [&](size_t bytes) {
        void* p = ws + o;
        o += (bytes + 255) & ~(size_t)255;
        return p;
    };
    __bf16* x_bf  = (__bf16*)alloc((size_t)B_SZ * IN_SZ * 2);
    __bf16* h0    = (__bf16*)alloc((size_t)B_SZ * RH_SZ * 2);
    __bf16* h1    = (__bf16*)alloc((size_t)B_SZ * RH_SZ * 2);
    int*    choice= (int*)   alloc((size_t)B_SZ * 4);
    int*    idx   = (int*)   alloc((size_t)B_SZ * 4);
    int*    meta  = (int*)   alloc(256);
    __bf16* t1    = (__bf16*)alloc((size_t)B_SZ * H_SZ * 2);
    __bf16* t2    = (__bf16*)alloc((size_t)B_SZ * H_SZ * 2);
    __bf16* rw0t  = (__bf16*)alloc((size_t)IN_SZ * RH_SZ * 2);
    __bf16* rw1t  = (__bf16*)alloc((size_t)RH_SZ * RH_SZ * 2);
    __bf16* w1t   = (__bf16*)alloc((size_t)E_SZ * IN_SZ * H_SZ * 2);
    __bf16* w2t   = (__bf16*)alloc((size_t)E_SZ * H_SZ * H_SZ * 2);
    __bf16* w3t   = (__bf16*)alloc((size_t)E_SZ * H_SZ * OUT_SZ * 2);
    (void)ws_size; (void)in_sizes; (void)n_in; (void)out_size;

    // 0) weight prep: fp32 [K,N] -> bf16 [N,K] per expert (one-shot per call)
    transpose_cvt_kernel<<<dim3(IN_SZ / 64, RH_SZ / 64, 1), 256, 0, stream>>>(rw0, rw0t, IN_SZ, RH_SZ);
    transpose_cvt_kernel<<<dim3(RH_SZ / 64, RH_SZ / 64, 1), 256, 0, stream>>>(rw1, rw1t, RH_SZ, RH_SZ);
    transpose_cvt_kernel<<<dim3(IN_SZ / 64, H_SZ / 64, E_SZ), 256, 0, stream>>>(w1, w1t, IN_SZ, H_SZ);
    transpose_cvt_kernel<<<dim3(H_SZ / 64, H_SZ / 64, E_SZ), 256, 0, stream>>>(w2, w2t, H_SZ, H_SZ);
    transpose_cvt_kernel<<<dim3(H_SZ / 64, OUT_SZ / 64, E_SZ), 256, 0, stream>>>(w3, w3t, H_SZ, OUT_SZ);

    // 1) x -> bf16
    cvt_kernel<<<(B_SZ * IN_SZ) / (256 * 4), 256, 0, stream>>>(x, x_bf, B_SZ * IN_SZ);

    // 2) Router GEMM 0:  h0 = x @ rw0 + rb0   (no activation)
    gemm_bf16_wmma<false, false, false><<<dim3(B_SZ / BM, RH_SZ / BN, 1), 256, 0, stream>>>(
        x_bf, rw0t, rb0, h0, nullptr, nullptr, nullptr, B_SZ, RH_SZ, IN_SZ);

    // 3) Router GEMM 1:  h1 = elu(h0 @ rw1 + rb1)
    gemm_bf16_wmma<true, false, false><<<dim3(B_SZ / BM, RH_SZ / BN, 1), 256, 0, stream>>>(
        h0, rw1t, rb1, h1, nullptr, nullptr, nullptr, B_SZ, RH_SZ, RH_SZ);

    // 4) logits + argmax
    router_argmax<<<B_SZ / 8, 256, 0, stream>>>(h1, rw2, rb2, choice);

    // 5) build per-expert contiguous segments
    zero_meta<<<1, 32, 0, stream>>>(meta);
    count_kernel<<<B_SZ / 256, 256, 0, stream>>>(choice, meta);
    scan_kernel<<<1, 32, 0, stream>>>(meta);
    scatter_kernel<<<B_SZ / 256, 256, 0, stream>>>(choice, meta, idx);

    const int* seg = meta + 8;

    // 6) Expert GEMM 1: t1 = gather(x) @ w1[e] + b1[e]   (no activation)
    gemm_bf16_wmma<false, true, false><<<dim3(B_SZ / BM, H_SZ / BN, E_SZ), 256, 0, stream>>>(
        x_bf, w1t, b1, t1, nullptr, seg, idx, B_SZ, H_SZ, IN_SZ);

    // 7) Expert GEMM 2: t2 = elu(t1 @ w2[e] + b2[e])
    gemm_bf16_wmma<true, false, false><<<dim3(B_SZ / BM, H_SZ / BN, E_SZ), 256, 0, stream>>>(
        t1, w2t, b2, t2, nullptr, seg, idx, B_SZ, H_SZ, H_SZ);

    // 8) Expert GEMM 3: out[scatter] = t2 @ w3[e] + b3[e]
    gemm_bf16_wmma<false, false, true><<<dim3(B_SZ / BM, OUT_SZ / BN, E_SZ), 256, 0, stream>>>(
        t2, w3t, b3, nullptr, out, seg, idx, B_SZ, OUT_SZ, H_SZ);
}
